// TransformerClassification_3530463117866
// MI455X (gfx1250) — matrix operand order repaired and checked by reading, not compiled, not run on hardware
//
#include <hip/hip_runtime.h>

#define LNUM 6
#define HNUM 12
#define DD   49
#define TT   16
#define DP   64
#define DFFN 1024
#define OUTC 10
#define BB   8192
#define BT   (BB*TT)

typedef __attribute__((ext_vector_type(16))) _Float16 v16h;
typedef __attribute__((ext_vector_type(8)))  _Float16 v8h;
typedef __attribute__((ext_vector_type(8)))  float    v8f;

// ---------------------------------------------------------------- WMMA utils
__device__ __forceinline__ v8f wmma32(v16h a, v16h b, v8f c) {
  return __builtin_amdgcn_wmma_f32_16x16x32_f16(false, a, false, b, (short)0, c,
                                                false, false);
}

// A-matrix 16x32 f16 fragment from a row-major buffer (ld halves per row).
// lanes 0-15: M=lane, K {0..15}; lanes 16-31: M=lane-16, K {16..31}
// Each lane = two contiguous 16-byte runs -> 2 vector loads.
__device__ __forceinline__ v16h load_a_h(const _Float16* __restrict__ p, int ld,
                                         int k0, int lane) {
  int m = lane & 15;
  int kb = (lane >> 4) * 16;
  const _Float16* r = p + (size_t)m * ld + k0 + kb;
  v8h lo = *(const v8h*)(r);
  v8h hi = *(const v8h*)(r + 8);
  return __builtin_shufflevector(lo, hi, 0, 1, 2, 3, 4, 5, 6, 7, 8, 9, 10, 11,
                                 12, 13, 14, 15);
}

// Same A layout but fp32 source (converted on the fly), vectorized float4.
__device__ __forceinline__ v16h load_a_f(const float* __restrict__ p, int ld,
                                         int k0, int lane) {
  int m = lane & 15;
  int kb = (lane >> 4) * 16;
  const float* r = p + (size_t)m * ld + k0 + kb;
  float4 a0 = *(const float4*)(r);
  float4 a1 = *(const float4*)(r + 4);
  float4 b0 = *(const float4*)(r + 8);
  float4 b1 = *(const float4*)(r + 12);
  v16h a;
  a[0] = (_Float16)a0.x;  a[1] = (_Float16)a0.y;
  a[2] = (_Float16)a0.z;  a[3] = (_Float16)a0.w;
  a[4] = (_Float16)a1.x;  a[5] = (_Float16)a1.y;
  a[6] = (_Float16)a1.z;  a[7] = (_Float16)a1.w;
  a[8] = (_Float16)b0.x;  a[9] = (_Float16)b0.y;
  a[10] = (_Float16)b0.z; a[11] = (_Float16)b0.w;
  a[12] = (_Float16)b1.x; a[13] = (_Float16)b1.y;
  a[14] = (_Float16)b1.z; a[15] = (_Float16)b1.w;
  return a;
}

// B-matrix 32x16 fragment from FRAGMENT-PACKED storage: one contiguous
// 32-byte run per lane -> 2x b128 loads, fully coalesced across the wave.
// Packed layout: dst[(((kt*ntiles + nt)*32) + lane)*16 + i]
__device__ __forceinline__ v16h load_b_pk(const _Float16* __restrict__ p,
                                          int ntiles, int kt, int nt,
                                          int lane) {
  const _Float16* r = p + (((size_t)kt * ntiles + nt) * 32 + lane) * 16;
  return *(const v16h*)r;
}

// ---------------------------------------------------------------- prep kernels
// Pack an (R x C) fp32 matrix into fragment-packed, zero-padded (Kp x Np) fp16.
__global__ __launch_bounds__(256) void pack_b_k(const float* __restrict__ src,
                                                _Float16* __restrict__ dst,
                                                int R, int C, int Kp, int Np,
                                                int nmat) {
  int idx = blockIdx.x * blockDim.x + threadIdx.x;
  int per = Kp * Np;
  if (idx >= nmat * per) return;
  int m = idx / per;
  int rem = idx % per;
  int i = rem & 15;
  int lane = (rem >> 4) & 31;
  int tile = rem >> 9;
  int ntiles = Np >> 4;
  int kt = tile / ntiles, nt = tile % ntiles;
  int K = kt * 32 + ((lane >> 4) << 4) + i;
  int N = nt * 16 + (lane & 15);
  float v = (K < R && N < C) ? src[((size_t)m * R + K) * C + N] : 0.f;
  dst[idx] = (_Float16)v;
}

// Wfold[l,h] = Wo[l,h] @ Wcat[l][h*49:(h+1)*49,:], emitted DIRECTLY in packed
// fragment layout (Kp=64 -> 2 ktiles, Np=64 -> 4 ntiles).
__global__ __launch_bounds__(256) void fold_w_k(const float* __restrict__ Wo,
                                                const float* __restrict__ Wcat,
                                                _Float16* __restrict__ Wfold) {
  int idx = blockIdx.x * blockDim.x + threadIdx.x;
  if (idx >= LNUM * HNUM * DP * DP) return;
  int i = idx & 15;
  int lane = (idx >> 4) & 31;
  int tile = (idx >> 9) & 7;   // kt*4 + nt
  int lh = idx >> 12;
  int nt = tile & 3, kt = tile >> 2;
  int K = kt * 32 + ((lane >> 4) << 4) + i;  // dd (input row of Wfold)
  int N = nt * 16 + (lane & 15);             // f  (output col)
  int l = lh / HNUM, hh = lh % HNUM;
  float s = 0.f;
  if (K < DD && N < DD) {
    const float* wo = Wo + ((size_t)lh * DD + K) * DD;
    const float* wc = Wcat + ((size_t)l * HNUM * DD + hh * DD) * DD + N;
    for (int e = 0; e < DD; ++e) s += wo[e] * wc[(size_t)e * DD];
  }
  Wfold[idx] = (_Float16)s;
}

// bfold[l] = bcat[l] + sum_h bo[l,h] @ Wcat_h   (padded to 64, pads = 0)
__global__ __launch_bounds__(256) void fold_b_k(const float* __restrict__ bo,
                                                const float* __restrict__ Wcat,
                                                const float* __restrict__ bcat,
                                                float* __restrict__ bfold) {
  int idx = blockIdx.x * blockDim.x + threadIdx.x;
  if (idx >= LNUM * DP) return;
  int f = idx & 63, l = idx >> 6;
  float s = 0.f;
  if (f < DD) {
    s = bcat[l * DD + f];
    for (int hh = 0; hh < HNUM; ++hh)
      for (int e = 0; e < DD; ++e)
        s += bo[(l * HNUM + hh) * DD + e] *
             Wcat[((size_t)l * HNUM * DD + hh * DD + e) * DD + f];
  }
  bfold[idx] = s;
}

// h0[b*T+t][c] = x[b*784 + t*49 + c] for c<49 else 0
__global__ __launch_bounds__(256) void init_h_k(const float* __restrict__ x,
                                                float* __restrict__ h) {
  int idx = blockIdx.x * blockDim.x + threadIdx.x;
  if (idx >= BT * DP) return;
  int rr = idx >> 6;
  int c = idx & 63;
  int b = rr >> 4, t = rr & 15;
  h[idx] = (c < DD) ? x[(size_t)b * 784 + t * DD + c] : 0.f;
}

// ---------------------------------------------------------------- LayerNorm 1
__global__ __launch_bounds__(256) void ln_k(const float* __restrict__ h,
                                            const float* __restrict__ g,
                                            const float* __restrict__ bln,
                                            _Float16* __restrict__ xn,
                                            int layer) {
  int row = blockIdx.x * 8 + (threadIdx.x >> 5);
  int lane = threadIdx.x & 31;
  const float* r = h + (size_t)row * DP;
  float v0 = r[lane];
  bool has1 = (lane + 32) < DD;
  float v1 = has1 ? r[lane + 32] : 0.f;
  float s = v0 + v1;
#pragma unroll
  for (int m = 16; m >= 1; m >>= 1) s += __shfl_xor(s, m, 32);
  float mu = s * (1.f / DD);
  float d0 = v0 - mu;
  float d1 = has1 ? (v1 - mu) : 0.f;
  float q = d0 * d0 + d1 * d1;
#pragma unroll
  for (int m = 16; m >= 1; m >>= 1) q += __shfl_xor(q, m, 32);
  float rstd = rsqrtf(q * (1.f / DD) + 1e-5f);
  const float* gl = g + layer * DD;
  const float* bl = bln + layer * DD;
  _Float16* orow = xn + (size_t)row * DP;
  orow[lane] = (_Float16)(d0 * rstd * gl[lane] + bl[lane]);
  float o1 = 0.f;
  if (has1) o1 = d1 * rstd * gl[lane + 32] + bl[lane + 32];
  orow[lane + 32] = (_Float16)o1;
}

// ------------------------------------------------------- fused attention layer
// grid = B, block = 128 (4 waves); wave w handles heads 3w..3w+2
__global__ __launch_bounds__(128) void attn_k(
    const _Float16* __restrict__ xn, const float* __restrict__ hres,
    const _Float16* __restrict__ WqH, const _Float16* __restrict__ WkH,
    const _Float16* __restrict__ WvH, const float* __restrict__ bq,
    const float* __restrict__ bk, const float* __restrict__ bv,
    const _Float16* __restrict__ Wfold, const float* __restrict__ bfold,
    float* __restrict__ x2, int layer) {
  __shared__ __align__(32) _Float16 sQ[4][TT * DP];    // q tile / o tile (A)
  __shared__ __align__(32) _Float16 sKTp[4][2 * 32 * 16];  // k^T, B-frag packed
  __shared__ __align__(32) _Float16 sVp[4][4 * 32 * 16];   // v, B-frag packed
  __shared__ __align__(32) _Float16 sW[4][TT * 32];        // softmax w (A)
  __shared__ __align__(16) float sRed[TT * DP];

  int tid = threadIdx.x, lane = tid & 31, w = tid >> 5;
  int n = lane & 15, mb = (lane >> 4) * 8;
  size_t boff = (size_t)blockIdx.x * (TT * DP);

  for (int i = tid; i < TT * DP; i += 128) sRed[i] = 0.f;
  // v fragment rows 16..31 (the K zero-padding) live in lanes 16..31 slots
  for (int nt = 0; nt < 4; ++nt)
    for (int i = lane; i < 16 * 16; i += 32)
      sVp[w][(nt * 32 + 16) * 16 + i] = (_Float16)0.f;
  // softmax A-tile cols 16..31 zero (K padding)
  for (int i = lane; i < TT * 16; i += 32) {
    int t = i >> 4, c = 16 + (i & 15);
    sW[w][t * 32 + c] = (_Float16)0.f;
  }

  const _Float16* xt = xn + boff;
  v16h xa0 = load_a_h(xt, DP, 0, lane);
  v16h xa1 = load_a_h(xt, DP, 32, lane);
  v8f pz = {0.f, 0.f, 0.f, 0.f, 0.f, 0.f, 0.f, 0.f};
  v8f pacc[4] = {pz, pz, pz, pz};
  __syncthreads();

  for (int hi = 0; hi < 3; ++hi) {
    int head = w * 3 + hi;
    size_t widx = ((size_t)layer * HNUM + head) * (DP * DP);
    const _Float16* Wq = WqH + widx;
    const _Float16* Wk = WkH + widx;
    const _Float16* Wv = WvH + widx;
    size_t bidx = ((size_t)layer * HNUM + head) * DD;
    const float* bqh = bq + bidx;
    const float* bkh = bk + bidx;
    const float* bvh = bv + bidx;

    // prefetch next head's weight fragments (global_prefetch_b8);
    // each lane covers a 256B slice -> whole 8KB matrix per instruction
    if (hi < 2) {
      size_t nwidx = widx + (size_t)DP * DP;
      __builtin_prefetch(WqH + nwidx + lane * 128, 0, 1);
      __builtin_prefetch(WkH + nwidx + lane * 128, 0, 1);
      __builtin_prefetch(WvH + nwidx + lane * 128, 0, 1);
      __builtin_prefetch(Wfold + nwidx + lane * 128, 0, 1);
    }

#pragma unroll
    for (int nt = 0; nt < 4; ++nt) {
      int col = nt * 16 + n;
      float bqv = (col < DD) ? bqh[col] : 0.f;
      float bkv = (col < DD) ? bkh[col] : 0.f;
      float bvv = (col < DD) ? bvh[col] : 0.f;
      v8f qa = pz, ka = pz, va = pz;
      qa = wmma32(xa0, load_b_pk(Wq, 4, 0, nt, lane), qa);
      qa = wmma32(xa1, load_b_pk(Wq, 4, 1, nt, lane), qa);
      ka = wmma32(xa0, load_b_pk(Wk, 4, 0, nt, lane), ka);
      ka = wmma32(xa1, load_b_pk(Wk, 4, 1, nt, lane), ka);
      va = wmma32(xa0, load_b_pk(Wv, 4, 0, nt, lane), va);
      va = wmma32(xa1, load_b_pk(Wv, 4, 1, nt, lane), va);
#pragma unroll
      for (int i = 0; i < 8; ++i) {
        int t = mb + i;
        sQ[w][t * DP + col] = (_Float16)(qa[i] + bqv);
        // k^T fragment-packed: element (K=e=col, N=t):
        //   kt = col>>5, lane' = ((col>>4)&1)*16 + t, i' = col&15
        sKTp[w][(((col >> 5) * 32) + ((col >> 4) & 1) * 16 + t) * 16 +
                (col & 15)] = (_Float16)(ka[i] + bkv);
      }
      // v fragment-packed: element (K=s=t (<16), N=e=col):
      //   nt' = col>>4, lane' = col&15, i' = t  -> i runs contiguous: b128 store
      v8h vv;
#pragma unroll
      for (int i = 0; i < 8; ++i) vv[i] = (_Float16)(va[i] + bvv);
      *(v8h*)(&sVp[w][((col >> 4) * 32 + (col & 15)) * 16 + mb]) = vv;
    }
    __syncthreads();

    // scores = q @ k^T   (16x16, K = 64 padded dim)
    v16h qa0 = load_a_h(sQ[w], DP, 0, lane);
    v16h qa1 = load_a_h(sQ[w], DP, 32, lane);
    v8f sc = pz;
    sc = wmma32(qa0, load_b_pk(sKTp[w], 1, 0, 0, lane), sc);
    sc = wmma32(qa1, load_b_pk(sKTp[w], 1, 1, 0, lane), sc);

    // softmax over s (16 cols spread across a 16-lane half)
    const float scale = 1.f / 7.f;  // 1/sqrt(49)
#pragma unroll
    for (int i = 0; i < 8; ++i) {
      float v = sc[i] * scale;
      float m = v;
      m = fmaxf(m, __shfl_xor(m, 1, 32));
      m = fmaxf(m, __shfl_xor(m, 2, 32));
      m = fmaxf(m, __shfl_xor(m, 4, 32));
      m = fmaxf(m, __shfl_xor(m, 8, 32));
      float e = __expf(v - m);
      float ss = e;
      ss += __shfl_xor(ss, 1, 32);
      ss += __shfl_xor(ss, 2, 32);
      ss += __shfl_xor(ss, 4, 32);
      ss += __shfl_xor(ss, 8, 32);
      sW[w][(mb + i) * 32 + n] = (_Float16)(e / ss);
    }
    __syncthreads();

    // o = w @ v   (K = 16 real, zero-padded to 32); overwrite sQ with o
    v16h wa = load_a_h(sW[w], 32, 0, lane);
#pragma unroll
    for (int nt = 0; nt < 4; ++nt) {
      v8f oa = pz;
      oa = wmma32(wa, load_b_pk(sVp[w], 4, 0, nt, lane), oa);
#pragma unroll
      for (int i = 0; i < 8; ++i)
        sQ[w][(mb + i) * DP + nt * 16 + n] = (_Float16)oa[i];
    }
    __syncthreads();

    // pacc += o @ Wfold_h
    const _Float16* Wf = Wfold + widx;
    v16h oa0 = load_a_h(sQ[w], DP, 0, lane);
    v16h oa1 = load_a_h(sQ[w], DP, 32, lane);
#pragma unroll
    for (int nt = 0; nt < 4; ++nt) {
      pacc[nt] = wmma32(oa0, load_b_pk(Wf, 4, 0, nt, lane), pacc[nt]);
      pacc[nt] = wmma32(oa1, load_b_pk(Wf, 4, 1, nt, lane), pacc[nt]);
    }
    __syncthreads();
  }

  // cross-wave head reduction in LDS, then residual + folded-bias add
#pragma unroll
  for (int nt = 0; nt < 4; ++nt)
#pragma unroll
    for (int i = 0; i < 8; ++i)
      atomicAdd(&sRed[(mb + i) * DP + nt * 16 + n], pacc[nt][i]);
  __syncthreads();
  const float4* hr4 = (const float4*)(hres + boff);
  const float4* sr4 = (const float4*)sRed;
  const float4* bf4 = (const float4*)(bfold + layer * DP);
  float4* x24 = (float4*)(x2 + boff);
  for (int i = tid; i < (TT * DP) / 4; i += 128) {
    float4 a = hr4[i], r = sr4[i], b = bf4[i & 15];
    float4 o;
    o.x = a.x + r.x + b.x;
    o.y = a.y + r.y + b.y;
    o.z = a.z + r.z + b.z;
    o.w = a.w + r.w + b.w;
    x24[i] = o;
  }
}

// ----------------------------------------------------- fused LN2 + FFN layer
// grid = B, block = 256 (8 waves)
__global__ __launch_bounds__(256) void ff_k(
    const float* __restrict__ x2, const float* __restrict__ g,
    const float* __restrict__ bln, const _Float16* __restrict__ W1h,
    const float* __restrict__ b1, const _Float16* __restrict__ W2h,
    const float* __restrict__ b2, float* __restrict__ hout, int layer) {
  __shared__ __align__(32) _Float16 sXN[TT * DP];    // 2 KB
  __shared__ __align__(32) _Float16 sA1[TT * DFFN];  // 32 KB (A-tile for FF2)
  __shared__ __align__(16) float sPart[4][TT * 16];  // 4 KB

  int tid = threadIdx.x, lane = tid & 31, w = tid >> 5;
  int n = lane & 15, mb = (lane >> 4) * 8;
  size_t boff = (size_t)blockIdx.x * (TT * DP);
  const float* xr = x2 + boff;

  // LayerNorm2 (16 rows, one thread each)
  if (tid < TT) {
    const float* r = xr + tid * DP;
    float s = 0.f;
    for (int e = 0; e < DD; ++e) s += r[e];
    float mu = s * (1.f / DD);
    float q = 0.f;
    for (int e = 0; e < DD; ++e) { float d = r[e] - mu; q += d * d; }
    float rstd = rsqrtf(q * (1.f / DD) + 1e-5f);
    const float* gl = g + layer * DD;
    const float* bl = bln + layer * DD;
    for (int e = 0; e < DD; ++e)
      sXN[tid * DP + e] = (_Float16)((r[e] - mu) * rstd * gl[e] + bl[e]);
    for (int e = DD; e < DP; ++e) sXN[tid * DP + e] = (_Float16)0.f;
  }
  __syncthreads();

  // FF1: (16x64) @ (64x1024), ReLU -> LDS fp16 (row-major, consumed as A)
  v16h xa0 = load_a_h(sXN, DP, 0, lane);
  v16h xa1 = load_a_h(sXN, DP, 32, lane);
  const _Float16* W1l = W1h + (size_t)layer * DP * DFFN;
  v8f pz = {0.f, 0.f, 0.f, 0.f, 0.f, 0.f, 0.f, 0.f};
#pragma unroll
  for (int j = 0; j < 8; ++j) {
    int ntile = w * 8 + j;
    int n0 = ntile * 16;
    v8f acc = pz;
    acc = wmma32(xa0, load_b_pk(W1l, 64, 0, ntile, lane), acc);
    acc = wmma32(xa1, load_b_pk(W1l, 64, 1, ntile, lane), acc);
    float bb = b1[layer * DFFN + n0 + n];
#pragma unroll
    for (int i = 0; i < 8; ++i) {
      float t = fmaxf(acc[i] + bb, 0.f);
      sA1[(mb + i) * DFFN + n0 + n] = (_Float16)t;
    }
  }
  __syncthreads();

  // FF2: (16x1024) @ (1024x64); waves split: nt = w&3, K-half = w>>2
  int nt = w & 3, kh = w >> 2;
  const _Float16* W2l = W2h + (size_t)layer * DFFN * DP;
  v8f acc = pz;
#pragma unroll
  for (int ks = 0; ks < 16; ++ks) {
    int kt = kh * 16 + ks;
    acc = wmma32(load_a_h(sA1, DFFN, kt * 32, lane),
                 load_b_pk(W2l, 4, kt, nt, lane), acc);
  }
  if (kh == 1) {
#pragma unroll
    for (int i = 0; i < 8; ++i) sPart[nt][(mb + i) * 16 + n] = acc[i];
  }
  __syncthreads();
  if (kh == 0) {
    int col = nt * 16 + n;
    float bb = (col < DD) ? b2[layer * DD + col] : 0.f;
#pragma unroll
    for (int i = 0; i < 8; ++i) {
      int idx = (mb + i) * DP + col;
      hout[boff + idx] = acc[i] + sPart[nt][(mb + i) * 16 + n] + xr[idx] + bb;
    }
  }
}

// ------------------------------------------------------- classification head
// grid = B/16, one wave: rows = 16 batch elements' token-0 vectors
__global__ __launch_bounds__(32) void head_k(const float* __restrict__ hA,
                                             const _Float16* __restrict__ Wch,
                                             const float* __restrict__ bc,
                                             float* __restrict__ out) {
  int lane = threadIdx.x & 31;
  int b0 = blockIdx.x * 16;
  const float* base = hA + (size_t)b0 * (TT * DP);  // batch stride = T*DP
  v16h a0 = load_a_f(base, TT * DP, 0, lane);
  v16h a1 = load_a_f(base, TT * DP, 32, lane);
  v8f acc = {0.f, 0.f, 0.f, 0.f, 0.f, 0.f, 0.f, 0.f};
  acc = wmma32(a0, load_b_pk(Wch, 1, 0, 0, lane), acc);
  acc = wmma32(a1, load_b_pk(Wch, 1, 1, 0, lane), acc);
  int n = lane & 15, mb = (lane >> 4) * 8;
  if (n < OUTC) {
    float bcv = bc[n];
#pragma unroll
    for (int i = 0; i < 8; ++i)
      out[(size_t)(b0 + mb + i) * OUTC + n] = acc[i] + bcv;
  }
}

// ---------------------------------------------------------------- launcher
extern "C" void kernel_launch(void* const* d_in, const int* in_sizes, int n_in,
                              void* d_out, int out_size, void* d_ws,
                              size_t ws_size, hipStream_t stream) {
  (void)in_sizes; (void)n_in; (void)out_size; (void)ws_size;
  const float* x    = (const float*)d_in[0];
  const float* Wq   = (const float*)d_in[2];
  const float* bq   = (const float*)d_in[3];
  const float* Wk   = (const float*)d_in[4];
  const float* bk   = (const float*)d_in[5];
  const float* Wv   = (const float*)d_in[6];
  const float* bv   = (const float*)d_in[7];
  const float* Wo   = (const float*)d_in[8];
  const float* bo   = (const float*)d_in[9];
  const float* Wcat = (const float*)d_in[10];
  const float* bcat = (const float*)d_in[11];
  const float* ln1g = (const float*)d_in[12];
  const float* ln1b = (const float*)d_in[13];
  const float* ln2g = (const float*)d_in[14];
  const float* ln2b = (const float*)d_in[15];
  const float* W1   = (const float*)d_in[16];
  const float* b1   = (const float*)d_in[17];
  const float* W2   = (const float*)d_in[18];
  const float* b2   = (const float*)d_in[19];
  const float* Wc   = (const float*)d_in[20];
  const float* bc   = (const float*)d_in[21];
  float* out = (float*)d_out;

  char* p = (char*)d_ws;
  auto take = [&](size_t bytes) -> char* {
    char* r = p;
    p += (bytes + 255) & ~(size_t)255;
    return r;
  };
  float*    hA  = (float*)take((size_t)BT * DP * 4);
  float*    xx2 = (float*)take((size_t)BT * DP * 4);
  _Float16* xnb = (_Float16*)take((size_t)BT * DP * 2);
  _Float16* WqH = (_Float16*)take((size_t)LNUM * HNUM * DP * DP * 2);
  _Float16* WkH = (_Float16*)take((size_t)LNUM * HNUM * DP * DP * 2);
  _Float16* WvH = (_Float16*)take((size_t)LNUM * HNUM * DP * DP * 2);
  _Float16* Wfd = (_Float16*)take((size_t)LNUM * HNUM * DP * DP * 2);
  float*    bfd = (float*)take((size_t)LNUM * DP * 4);
  _Float16* W1h = (_Float16*)take((size_t)LNUM * DP * DFFN * 2);
  _Float16* W2h = (_Float16*)take((size_t)LNUM * DFFN * DP * 2);
  _Float16* Wch = (_Float16*)take((size_t)DP * 16 * 2);

  // weight prep: fp32 -> fp16, zero-padded, fragment-packed for b128 loads
  int nw = LNUM * HNUM * DP * DP;
  pack_b_k<<<(nw + 255) / 256, 256, 0, stream>>>(Wq, WqH, DD, DD, DP, DP,
                                                 LNUM * HNUM);
  pack_b_k<<<(nw + 255) / 256, 256, 0, stream>>>(Wk, WkH, DD, DD, DP, DP,
                                                 LNUM * HNUM);
  pack_b_k<<<(nw + 255) / 256, 256, 0, stream>>>(Wv, WvH, DD, DD, DP, DP,
                                                 LNUM * HNUM);
  int n1 = LNUM * DP * DFFN;
  pack_b_k<<<(n1 + 255) / 256, 256, 0, stream>>>(W1, W1h, DD, DFFN, DP, DFFN,
                                                 LNUM);
  int n2 = LNUM * DFFN * DP;
  pack_b_k<<<(n2 + 255) / 256, 256, 0, stream>>>(W2, W2h, DFFN, DD, DFFN, DP,
                                                 LNUM);
  pack_b_k<<<4, 256, 0, stream>>>(Wc, Wch, DD, OUTC, DP, 16, 1);
  fold_w_k<<<(nw + 255) / 256, 256, 0, stream>>>(Wo, Wcat, Wfd);
  fold_b_k<<<2, 256, 0, stream>>>(bo, Wcat, bcat, bfd);
  init_h_k<<<(BT * DP + 255) / 256, 256, 0, stream>>>(x, hA);

  for (int l = 0; l < LNUM; ++l) {
    ln_k<<<BT / 8, 256, 0, stream>>>(hA, ln1g, ln1b, xnb, l);
    attn_k<<<BB, 128, 0, stream>>>(xnb, hA, WqH, WkH, WvH, bq, bk, bv, Wfd, bfd,
                                   xx2, l);
    ff_k<<<BB, 256, 0, stream>>>(xx2, ln2g, ln2b, W1h, b1, W2h, b2, hA, l);
  }
  head_k<<<BB / 16, 32, 0, stream>>>(hA, Wch, bc, out);
}
